// GPCNDTA_12627203850543
// MI455X (gfx1250) — matrix operand order repaired and checked by reading, not compiled, not run on hardware
//
#include <hip/hip_runtime.h>

// Problem dims (fixed by the reference)
#define NN 2048   // N nodes
#define NE 4096   // E edges
#define DFV 64
#define DFE 16
#define DNH 128
#define DNC 128

typedef __attribute__((ext_vector_type(16))) __bf16 v16bf;
typedef __attribute__((ext_vector_type(8)))  float  v8f;

struct alignas(16) U4 { unsigned int x, y, z, w; };
struct U4x2 { U4 lo, hi; };
struct H8 { unsigned short h[8]; };

__device__ __forceinline__ float bf2f(__bf16 v) {
  unsigned short s = __builtin_bit_cast(unsigned short, v);
  unsigned int u = (unsigned int)s << 16;
  return __builtin_bit_cast(float, u);
}
__device__ __forceinline__ __bf16 f2bf(float f) {
  unsigned int u = __builtin_bit_cast(unsigned int, f);
  unsigned int r = u + 0x7FFFu + ((u >> 16) & 1u);
  return __builtin_bit_cast(__bf16, (unsigned short)(r >> 16));
}

// ---------------------------------------------------------------------------
// Tiled bf16 WMMA GEMM: D = A[MxK] * B[KxN] (+ fused epilogues)
//   mode 0: outF = acc
//   mode 1: v = (gr==gc ? 1 : acc) * maskF[gr,gc]; outB = bf16(v)   (gated adj)
//   mode 2: v = relu(acc + bias[gc]); outF = v; if(outB) outB = bf16(v)
//   mode 3: outF = acc + bias[gc]
//   mode 4: outB = bf16(acc)
// Block = 128 threads (4 waves), 64x64 tile, K-stage = 64 (2 WMMA sub-steps).
// A tile is staged global->LDS with GLOBAL_LOAD_ASYNC_TO_LDS_B128 (ASYNCcnt);
// B tile is staged transposed so fragments are contiguous ds_load_b128 reads.
// ---------------------------------------------------------------------------
__global__ __launch_bounds__(128)
void gemm_wmma_bf16(const __bf16* __restrict__ A, int lda,
                    const __bf16* __restrict__ B, int ldb,
                    int M, int N, int K,
                    float* __restrict__ outF, int ldof,
                    __bf16* __restrict__ outB, int ldob,
                    const float* __restrict__ maskF, int ldm,
                    const float* __restrict__ bias, int mode)
{
  __shared__ __bf16 sA[64][72];   // 64 rows x 64 K (pad 8, row = 144B, 16B-aligned)
  __shared__ __bf16 sB[64][72];   // transposed: sB[n][k]

  const int tid  = threadIdx.x;
  const int lane = tid & 31;
  const int wav  = tid >> 5;
  const int bm   = blockIdx.y * 64;
  const int bn   = blockIdx.x * 64;
  const int wm   = (wav >> 1) * 32;
  const int wn   = (wav & 1) * 32;
  const int l15  = lane & 15;
  const bool hi  = lane >= 16;

  const v8f vz = {0.f,0.f,0.f,0.f,0.f,0.f,0.f,0.f};
  v8f acc[2][2] = {{vz, vz}, {vz, vz}};

  for (int k0 = 0; k0 < K; k0 += 64) {
    #pragma unroll
    for (int it = 0; it < 4; ++it) {
      int c = tid + it * 128;                 // 512 16B chunks per operand tile
      { // A tile 64x64, row-major: async DMA straight into LDS (no VGPR hop)
        int ar = c >> 3;
        int ak = (c & 7) << 3;
        int gr = bm + ar, gk = k0 + ak;
        unsigned lds = (unsigned)(uintptr_t)&sA[ar][ak];
        if (gr < M && gk < K) {
          const __bf16* gp = A + (size_t)gr * lda + gk;
          asm volatile("global_load_async_to_lds_b128 %0, %1, off"
                       :: "v"(lds), "v"(gp) : "memory");
        } else {
          U4 z = {0u,0u,0u,0u};
          *reinterpret_cast<U4*>(&sA[ar][ak]) = z;
        }
      }
      { // B tile 64x64, loaded row-major, stored transposed into sB[n][k]
        int bk = c >> 3;
        int bc = (c & 7) << 3;
        U4 val = {0u,0u,0u,0u};
        int gk = k0 + bk, gn = bn + bc;
        if (gk < K && gn < N)
          val = *reinterpret_cast<const U4*>(B + (size_t)gk * ldb + gn);
        H8 hh = __builtin_bit_cast(H8, val);
        #pragma unroll
        for (int j = 0; j < 8; ++j)
          sB[bc + j][bk] = __builtin_bit_cast(__bf16, hh.h[j]);
      }
    }
    asm volatile("s_wait_asynccnt 0x0" ::: "memory");  // A-tile DMA complete
    __syncthreads();

    #pragma unroll
    for (int ks = 0; ks < 64; ks += 32) {
      // A frag (16x32): lane<16 row l15, K {0..7,16..23}; lane>=16 K {8..15,24..31}
      // B frag (32x16): lane<16 col l15, K {0..7,16..23}; lane>=16 K {8..15,24..31}
      v16bf afr[2], bfr[2];
      const int aK = ks + (hi ? 8 : 0);
      const int bK = ks + (hi ? 8 : 0);
      #pragma unroll
      for (int mi = 0; mi < 2; ++mi) {
        const __bf16* p = &sA[wm + mi * 16 + l15][0];
        U4x2 q{ *reinterpret_cast<const U4*>(p + aK),
                *reinterpret_cast<const U4*>(p + aK + 16) };
        afr[mi] = __builtin_bit_cast(v16bf, q);
      }
      #pragma unroll
      for (int ni = 0; ni < 2; ++ni) {
        const __bf16* p = &sB[wn + ni * 16 + l15][0];
        U4x2 q{ *reinterpret_cast<const U4*>(p + bK),
                *reinterpret_cast<const U4*>(p + bK + 16) };
        bfr[ni] = __builtin_bit_cast(v16bf, q);
      }
      #pragma unroll
      for (int mi = 0; mi < 2; ++mi)
        #pragma unroll
        for (int ni = 0; ni < 2; ++ni)
          acc[mi][ni] = __builtin_amdgcn_wmma_f32_16x16x32_bf16(
              false, afr[mi], false, bfr[ni], (short)0, acc[mi][ni], false, false);
    }
    __syncthreads();
  }

  // Epilogue. C/D layout: VGPR r -> row (hi?8:0)+r, col = l15
  #pragma unroll
  for (int mi = 0; mi < 2; ++mi) {
    #pragma unroll
    for (int ni = 0; ni < 2; ++ni) {
      int gc = bn + wn + ni * 16 + l15;
      int rb = bm + wm + mi * 16 + (hi ? 8 : 0);
      if (gc >= N) continue;
      #pragma unroll
      for (int r = 0; r < 8; ++r) {
        int gr = rb + r;
        if (gr >= M) continue;
        float v = acc[mi][ni][r];
        if (mode == 1) {
          if (gr == gc) v = 1.0f;
          v *= maskF[(size_t)gr * ldm + gc];
          outB[(size_t)gr * ldob + gc] = f2bf(v);
        } else if (mode == 2) {
          v += bias[gc];
          v = v > 0.f ? v : 0.f;
          outF[(size_t)gr * ldof + gc] = v;
          if (outB) outB[(size_t)gr * ldob + gc] = f2bf(v);
        } else if (mode == 3) {
          outF[(size_t)gr * ldof + gc] = v + bias[gc];
        } else if (mode == 4) {
          outB[(size_t)gr * ldob + gc] = f2bf(v);
        } else {
          outF[(size_t)gr * ldof + gc] = v;
        }
      }
    }
  }
}

// ---------------------------------------------------------------------------
// Elementwise / small helper kernels
// ---------------------------------------------------------------------------
__global__ void cast_f2b(const float* __restrict__ in, __bf16* __restrict__ out,
                         long n, int relu)
{
  long i = (long)blockIdx.x * blockDim.x + threadIdx.x;
  long st = (long)gridDim.x * blockDim.x;
  for (; i < n; i += st) {
    float v = in[i];
    if (relu && v < 0.f) v = 0.f;
    out[i] = f2bf(v);
  }
}

// out[c*R + r] = bf16(in[r*C + c])
__global__ void transpose_cast(const float* __restrict__ in, __bf16* __restrict__ out,
                               int R, int C)
{
  __shared__ float t[32][33];
  int c0 = blockIdx.x * 32, r0 = blockIdx.y * 32;
  int tx = threadIdx.x, ty = threadIdx.y;     // block (32,8)
  #pragma unroll
  for (int j = 0; j < 32; j += 8) {
    int r = r0 + ty + j, c = c0 + tx;
    t[ty + j][tx] = (r < R && c < C) ? in[(size_t)r * C + c] : 0.f;
  }
  __syncthreads();
  #pragma unroll
  for (int j = 0; j < 32; j += 8) {
    int c = c0 + ty + j, r = r0 + tx;
    if (c < C && r < R) out[(size_t)c * R + r] = f2bf(t[tx][ty + j]);
  }
}

// s[row] = sum_d H[row,d] * p[d]
__global__ void gate_dot(const float* __restrict__ H, int ld,
                         const float* __restrict__ p, float* __restrict__ s,
                         int rows, int dims)
{
  int r = blockIdx.x * blockDim.x + threadIdx.x;
  if (r >= rows) return;
  float a = 0.f;
  for (int d = 0; d < dims; ++d) a += H[(size_t)r * ld + d] * p[d];
  s[r] = a;
}

// out[i] = bf16(Tf32[i] * s[i % cols])
__global__ void scale_cols_f32(const float* __restrict__ T, const float* __restrict__ s,
                               __bf16* __restrict__ out, long n, int cols)
{
  long i = (long)blockIdx.x * blockDim.x + threadIdx.x;
  long st = (long)gridDim.x * blockDim.x;
  for (; i < n; i += st) out[i] = f2bf(T[i] * s[i % cols]);
}

// out[i] = bf16(float(Tbf[i]) * s[i % cols])
__global__ void scale_cols_bf(const __bf16* __restrict__ T, const float* __restrict__ s,
                              __bf16* __restrict__ out, long n, int cols)
{
  long i = (long)blockIdx.x * blockDim.x + threadIdx.x;
  long st = (long)gridDim.x * blockDim.x;
  for (; i < n; i += st) out[i] = f2bf(bf2f(T[i]) * s[i % cols]);
}

// cmax[j] = max_i A[i*cols + j]
__global__ void col_max(const __bf16* __restrict__ A, float* __restrict__ cmax,
                        int rows, int cols)
{
  int j = blockIdx.x * blockDim.x + threadIdx.x;
  if (j >= cols) return;
  float m = -3.4e38f;
  for (int i = 0; i < rows; ++i) {
    float v = bf2f(A[(size_t)i * cols + j]);
    m = v > m ? v : m;
  }
  cmax[j] = m;
}

// out[r,c] = bf16(in[r,c] / cmax[r])   (fold column-max into the small operand)
__global__ void scale_rows_div(const float* __restrict__ in, const float* __restrict__ cm,
                               __bf16* __restrict__ out, long n, int cols)
{
  long i = (long)blockIdx.x * blockDim.x + threadIdx.x;
  long st = (long)gridDim.x * blockDim.x;
  for (; i < n; i += st) out[i] = f2bf(in[i] / cm[i / cols]);
}

// relu(layernorm(row)) dual-stored (f32 + bf16) into concat buffers
__global__ void layernorm_relu(const float* __restrict__ in, int ldin,
                               const float* __restrict__ g, const float* __restrict__ b,
                               float* __restrict__ outF, int ldof,
                               __bf16* __restrict__ outB, int ldob,
                               int rows, int D)
{
  int r = blockIdx.x * blockDim.x + threadIdx.x;
  if (r >= rows) return;
  const float* x = in + (size_t)r * ldin;
  float s = 0.f, s2 = 0.f;
  for (int d = 0; d < D; ++d) { float v = x[d]; s += v; s2 += v * v; }
  float m = s / D;
  float var = s2 / D - m * m;
  float inv = rsqrtf(var + 1e-5f);
  for (int d = 0; d < D; ++d) {
    float v = (x[d] - m) * inv * g[d] + b[d];
    if (v < 0.f) v = 0.f;
    outF[(size_t)r * ldof + d] = v;
    outB[(size_t)r * ldob + d] = f2bf(v);
  }
}

// ---------------------------------------------------------------------------
static void launch_gemm(hipStream_t st, const __bf16* A, int lda, const __bf16* B, int ldb,
                        int M, int N, int K, float* oF, int ldof, __bf16* oB, int ldob,
                        const float* mask, int ldm, const float* bias, int mode)
{
  dim3 grid((N + 63) / 64, (M + 63) / 64);
  gemm_wmma_bf16<<<grid, dim3(128), 0, st>>>(A, lda, B, ldb, M, N, K,
                                             oF, ldof, oB, ldob, mask, ldm, bias, mode);
}

extern "C" void kernel_launch(void* const* d_in, const int* in_sizes, int n_in,
                              void* d_out, int out_size, void* d_ws, size_t ws_size,
                              hipStream_t stream)
{
  (void)in_sizes; (void)n_in; (void)out_size; (void)ws_size;
  const float* X     = (const float*)d_in[0];
  const float* Z     = (const float*)d_in[1];
  const float* adj_e = (const float*)d_in[2];
  const float* adj_v = (const float*)d_in[3];
  const float* T     = (const float*)d_in[4];
  const float* W1 = (const float*)d_in[5];
  const float* p1 = (const float*)d_in[6];
  const float* b1 = (const float*)d_in[7];
  const float* Wf1= (const float*)d_in[8];
  const float* g1 = (const float*)d_in[9];
  const float* be1= (const float*)d_in[10];
  const float* W2 = (const float*)d_in[11];
  const float* p2 = (const float*)d_in[12];
  const float* b2 = (const float*)d_in[13];
  const float* Wf2= (const float*)d_in[14];
  const float* g2 = (const float*)d_in[15];
  const float* be2= (const float*)d_in[16];
  const float* W3 = (const float*)d_in[17];
  const float* p3 = (const float*)d_in[18];
  const float* b3 = (const float*)d_in[19];
  const float* W4 = (const float*)d_in[20];
  const float* p4 = (const float*)d_in[21];
  const float* b4 = (const float*)d_in[22];
  const float* W5 = (const float*)d_in[23];
  const float* p5 = (const float*)d_in[24];
  const float* b5 = (const float*)d_in[25];

  char* base = (char*)d_ws;
  size_t off = 0;
  auto wsAlloc = [&](size_t bytes) -> void* {
    void* p = base + off;
    off += (bytes + 255) & ~(size_t)255;
    return p;
  };

  __bf16* Tb    = (__bf16*)wsAlloc((size_t)NN * NE * 2);       // bf16 T        [N,E]
  __bf16* Ttb   = (__bf16*)wsAlloc((size_t)NE * NN * 2);       // bf16 T^T      [E,N]
  __bf16* Asc   = (__bf16*)wsAlloc((size_t)NE * NN * 2);       // gated operand
  __bf16* adjAb = (__bf16*)wsAlloc((size_t)NE * NE * 2);       // gated adjacency (bf16)
  __bf16* Xb    = (__bf16*)wsAlloc((size_t)NN * DFV * 2);
  __bf16* Zb    = (__bf16*)wsAlloc((size_t)NE * DFE * 2);
  __bf16* Z1b   = (__bf16*)wsAlloc((size_t)NE * DFE * 2);      // relu(Z)
  __bf16* W1b   = (__bf16*)wsAlloc((size_t)DFV * DNH * 2);
  __bf16* Wf1b  = (__bf16*)wsAlloc((size_t)DFV * DNH * 2);
  __bf16* W2b   = (__bf16*)wsAlloc((size_t)DFE * DFE * 2);
  __bf16* Wf2b  = (__bf16*)wsAlloc((size_t)DFE * DFE * 2);
  __bf16* W3b   = (__bf16*)wsAlloc((size_t)2 * DNH * DNH * 2);
  __bf16* W4b   = (__bf16*)wsAlloc((size_t)2 * DFE * DFE * 2);
  __bf16* W5b   = (__bf16*)wsAlloc((size_t)DNH * DNC * 2);
  __bf16* HWb   = (__bf16*)wsAlloc((size_t)NN * DNH * 2);      // small GEMM operand (bf16)
  float*  tmpF  = (float*) wsAlloc((size_t)NN * DNH * 4);      // small GEMM f32 result
  float*  X1F1  = (float*) wsAlloc((size_t)NN * 2 * DNH * 4);
  __bf16* X1F1b = (__bf16*)wsAlloc((size_t)NN * 2 * DNH * 2);
  float*  Z2F2  = (float*) wsAlloc((size_t)NE * 2 * DFE * 4);
  __bf16* Z2F2b = (__bf16*)wsAlloc((size_t)NE * 2 * DFE * 2);
  float*  X3    = (float*) wsAlloc((size_t)NN * DNH * 4);
  __bf16* X3b   = (__bf16*)wsAlloc((size_t)NN * DNH * 2);
  float*  Z4    = (float*) wsAlloc((size_t)NE * DFE * 4);
  float*  sv    = (float*) wsAlloc((size_t)NE * 4);
  float*  cmax  = (float*) wsAlloc((size_t)NE * 4);

  // ---- precompute bf16 casts ----
  cast_f2b<<<2048, 256, 0, stream>>>(T, Tb, (long)NN * NE, 0);
  transpose_cast<<<dim3(NE / 32, NN / 32), dim3(32, 8), 0, stream>>>(T, Ttb, NN, NE);
  cast_f2b<<<512, 256, 0, stream>>>(X, Xb, (long)NN * DFV, 0);
  cast_f2b<<<256, 256, 0, stream>>>(Z, Zb, (long)NE * DFE, 0);
  cast_f2b<<<256, 256, 0, stream>>>(Z, Z1b, (long)NE * DFE, 1);
  cast_f2b<<<32, 256, 0, stream>>>(W1,  W1b,  (long)DFV * DNH, 0);
  cast_f2b<<<32, 256, 0, stream>>>(Wf1, Wf1b, (long)DFV * DNH, 0);
  cast_f2b<<<1, 256, 0, stream>>>(W2,  W2b,  (long)DFE * DFE, 0);
  cast_f2b<<<1, 256, 0, stream>>>(Wf2, Wf2b, (long)DFE * DFE, 0);
  cast_f2b<<<128, 256, 0, stream>>>(W3, W3b, (long)2 * DNH * DNH, 0);
  cast_f2b<<<4, 256, 0, stream>>>(W4, W4b, (long)2 * DFE * DFE, 0);
  cast_f2b<<<64, 256, 0, stream>>>(W5, W5b, (long)DNH * DNC, 0);

  // ---- Layer 1 (node gc): X1 = relu(adjA1 @ (X@W1) + b1) ----
  gate_dot<<<(NE + 127) / 128, 128, 0, stream>>>(Z, DFE, p1, sv, NE, DFE);
  scale_cols_f32<<<2048, 256, 0, stream>>>(T, sv, Asc, (long)NN * NE, NE);
  launch_gemm(stream, Asc, NE, Ttb, NN, NN, NN, NE,
              nullptr, 0, adjAb, NN, adj_v, NN, nullptr, 1);
  launch_gemm(stream, Xb, DFV, W1b, DNH, NN, DNH, DFV,
              nullptr, 0, HWb, DNH, nullptr, 0, nullptr, 4);
  launch_gemm(stream, adjAb, NN, HWb, DNH, NN, DNH, NN,
              X1F1, 2 * DNH, X1F1b, 2 * DNH, nullptr, 0, b1, 2);
  // F1 = relu(LN(X@Wf1)) into right half of X1F1
  launch_gemm(stream, Xb, DFV, Wf1b, DNH, NN, DNH, DFV,
              tmpF, DNH, nullptr, 0, nullptr, 0, nullptr, 0);
  layernorm_relu<<<(NN + 127) / 128, 128, 0, stream>>>(tmpF, DNH, g1, be1,
      X1F1 + DNH, 2 * DNH, X1F1b + DNH, 2 * DNH, NN, DNH);

  // ---- Layer 2 (edge gc): Z2 = relu(norm(adjA2) @ (Z1@W2) + b2) ----
  gate_dot<<<(NN + 127) / 128, 128, 0, stream>>>(X1F1, 2 * DNH, p2, sv, NN, 2 * DNH);
  scale_cols_bf<<<2048, 256, 0, stream>>>(Ttb, sv, Asc, (long)NE * NN, NN);
  launch_gemm(stream, Asc, NN, Tb, NE, NE, NE, NN,
              nullptr, 0, adjAb, NE, adj_e, NE, nullptr, 1);
  launch_gemm(stream, Z1b, DFE, W2b, DFE, NE, DFE, DFE,
              tmpF, DFE, nullptr, 0, nullptr, 0, nullptr, 0);
  col_max<<<NE / 128, 128, 0, stream>>>(adjAb, cmax, NE, NE);
  scale_rows_div<<<256, 256, 0, stream>>>(tmpF, cmax, HWb, (long)NE * DFE, DFE);
  launch_gemm(stream, adjAb, NE, HWb, DFE, NE, DFE, NE,
              Z2F2, 2 * DFE, Z2F2b, 2 * DFE, nullptr, 0, b2, 2);
  // F2 = relu(LN(Z@Wf2)) into right half of Z2F2
  launch_gemm(stream, Zb, DFE, Wf2b, DFE, NE, DFE, DFE,
              tmpF, DFE, nullptr, 0, nullptr, 0, nullptr, 0);
  layernorm_relu<<<(NE + 127) / 128, 128, 0, stream>>>(tmpF, DFE, g2, be2,
      Z2F2 + DFE, 2 * DFE, Z2F2b + DFE, 2 * DFE, NE, DFE);

  // ---- Layer 3 (node gc): X3 = relu(adjA3 @ (X1F1@W3) + b3) ----
  gate_dot<<<(NE + 127) / 128, 128, 0, stream>>>(Z2F2, 2 * DFE, p3, sv, NE, 2 * DFE);
  scale_cols_f32<<<2048, 256, 0, stream>>>(T, sv, Asc, (long)NN * NE, NE);
  launch_gemm(stream, Asc, NE, Ttb, NN, NN, NN, NE,
              nullptr, 0, adjAb, NN, adj_v, NN, nullptr, 1);
  launch_gemm(stream, X1F1b, 2 * DNH, W3b, DNH, NN, DNH, 2 * DNH,
              nullptr, 0, HWb, DNH, nullptr, 0, nullptr, 4);
  launch_gemm(stream, adjAb, NN, HWb, DNH, NN, DNH, NN,
              X3, DNH, X3b, DNH, nullptr, 0, b3, 2);

  // ---- Layer 4 (edge gc): Z4 = relu(norm(adjA4) @ (Z2F2@W4) + b4) ----
  gate_dot<<<(NN + 127) / 128, 128, 0, stream>>>(X3, DNH, p4, sv, NN, DNH);
  scale_cols_bf<<<2048, 256, 0, stream>>>(Ttb, sv, Asc, (long)NE * NN, NN);
  launch_gemm(stream, Asc, NN, Tb, NE, NE, NE, NN,
              nullptr, 0, adjAb, NE, adj_e, NE, nullptr, 1);
  launch_gemm(stream, Z2F2b, 2 * DFE, W4b, DFE, NE, DFE, 2 * DFE,
              tmpF, DFE, nullptr, 0, nullptr, 0, nullptr, 0);
  col_max<<<NE / 128, 128, 0, stream>>>(adjAb, cmax, NE, NE);
  scale_rows_div<<<256, 256, 0, stream>>>(tmpF, cmax, HWb, (long)NE * DFE, DFE);
  launch_gemm(stream, adjAb, NE, HWb, DFE, NE, DFE, NE,
              Z4, DFE, nullptr, 0, nullptr, 0, b4, 2);

  // ---- Layer 5 (node gc, no relu): X5 = adjA5 @ (X3@W5) + b5 ----
  gate_dot<<<(NE + 127) / 128, 128, 0, stream>>>(Z4, DFE, p5, sv, NE, DFE);
  scale_cols_f32<<<2048, 256, 0, stream>>>(T, sv, Asc, (long)NN * NE, NE);
  launch_gemm(stream, Asc, NE, Ttb, NN, NN, NN, NE,
              nullptr, 0, adjAb, NN, adj_v, NN, nullptr, 1);
  launch_gemm(stream, X3b, DNH, W5b, DNC, NN, DNC, DNH,
              nullptr, 0, HWb, DNC, nullptr, 0, nullptr, 4);
  launch_gemm(stream, adjAb, NN, HWb, DNC, NN, DNC, NN,
              (float*)d_out, DNC, nullptr, 0, nullptr, 0, b5, 3);
}
